// GraphConv_56573309223701
// MI455X (gfx1250) — matrix operand order repaired and checked by reading, not compiled, not run on hardware
//
#include <hip/hip_runtime.h>

typedef __attribute__((ext_vector_type(16))) _Float16 v16h;
typedef __attribute__((ext_vector_type(8)))  float    v8f;

#define CH 128   // IN_C == OUT_C == 128

// ---------------------------------------------------------------------------
// GEMM: out[m, n] = sum_k x[m, k] * W[n, k] + b[n]
// Block = 16 rows of x. 8 waves per block; wave w computes the 16x16 tile at
// columns [16w, 16w+16). K loop: 4 x V_WMMA_F32_16X16X32_F16 (K=32 each).
// ---------------------------------------------------------------------------
__global__ __launch_bounds__(256) void gc_gemm_wmma(
    const float* __restrict__ x, const float* __restrict__ W,
    const float* __restrict__ b, float* __restrict__ out, int nrows)
{
    const int lane = threadIdx.x & 31;
    const int wave = threadIdx.x >> 5;        // 0..7 -> column tile
    const int half = lane >> 4;               // 0: lanes 0-15, 1: lanes 16-31
    const int l    = lane & 15;

    const int row_base = blockIdx.x * 16;
    const int col_base = wave * 16;

    int m = row_base + l;
    if (m >= nrows) m = nrows - 1;            // clamp reads (exact grid anyway)

    const float* xrow = x + (size_t)m * CH;                 // A row (M = l)
    const float* wrow = W + (size_t)(col_base + l) * CH;    // B column (N = l)

    // C init: broadcast bias over M; N = lane % 16 per C/D layout.
    const float bias = b[col_base + l];
    v8f acc;
#pragma unroll
    for (int r = 0; r < 8; ++r) acc[r] = bias;

#pragma unroll
    for (int kt = 0; kt < CH; kt += 32) {
        // ---- A fragment (16x32 f16): lanes 0-15 hold K {kt..kt+7, kt+16..kt+23}
        //                              lanes 16-31 hold K shifted by +8
        const int ka = kt + half * 8;
        v16h afrag;
#pragma unroll
        for (int i = 0; i < 8; ++i) {
            afrag[i]     = (_Float16)xrow[ka + i];
            afrag[8 + i] = (_Float16)xrow[ka + 16 + i];
        }
        // ---- B fragment (32x16 f16): lanes 0-15 hold K {kt..kt+7, kt+16..kt+23},
        //                              lanes 16-31 hold K shifted by +8 (col = l)
        const int kb = kt + half * 8;
        v16h bfrag;
#pragma unroll
        for (int i = 0; i < 8; ++i) {
            bfrag[i]     = (_Float16)wrow[kb + i];
            bfrag[8 + i] = (_Float16)wrow[kb + 16 + i];
        }

        acc = __builtin_amdgcn_wmma_f32_16x16x32_f16(
            /*neg_a=*/false, afrag, /*neg_b=*/false, bfrag,
            /*c_mod=*/(short)0, acc, /*reuse_a=*/false, /*reuse_b=*/false);
    }

    // ---- store: VGPR r -> M = r + 8*half, N = l
#pragma unroll
    for (int r = 0; r < 8; ++r) {
        const int mr = row_base + r + 8 * half;
        if (mr < nrows)
            out[(size_t)mr * CH + col_base + l] = acc[r];
    }
}

// ---------------------------------------------------------------------------
// cnt[i] = 0
// ---------------------------------------------------------------------------
__global__ void gc_zero_cnt(int* __restrict__ cnt, int n)
{
    int i = blockIdx.x * blockDim.x + threadIdx.x;
    if (i < n) cnt[i] = 0;
}

// ---------------------------------------------------------------------------
// cnt[row[e]] += 1   (duplicates accumulate, matching .at[row].add(1.0))
// ---------------------------------------------------------------------------
__global__ void gc_hist(const long long* __restrict__ row,
                        int* __restrict__ cnt, int n)
{
    int i = blockIdx.x * blockDim.x + threadIdx.x;
    if (i < n) atomicAdd(&cnt[(int)row[i]], 1);
}

// ---------------------------------------------------------------------------
// result[j,c] = out[j,c] + out[col[j],c] + (float)cnt[j]
// thread = one (j,c) element, c fastest -> fully coalesced.
// ---------------------------------------------------------------------------
__global__ void gc_combine(const float* __restrict__ out,
                           const long long* __restrict__ col,
                           const int* __restrict__ cnt,
                           float* __restrict__ res, int n)
{
    long long idx = (long long)blockIdx.x * blockDim.x + threadIdx.x;
    if (idx >= (long long)n * CH) return;
    const int j = (int)(idx >> 7);
    const int c = (int)(idx & (CH - 1));
    const long long cj = col[j];
    res[idx] = out[idx] + out[cj * CH + c] + (float)cnt[j];
}

// ---------------------------------------------------------------------------
// Inputs (setup_inputs order): x [N*128 f32], edge_index [2*N i64],
//                              W [128*128 f32], b [128 f32]
// Workspace layout: out (N*128 f32 = 256 MB) | cnt (N i32 = 2 MB)
// ---------------------------------------------------------------------------
extern "C" void kernel_launch(void* const* d_in, const int* in_sizes, int n_in,
                              void* d_out, int out_size, void* d_ws, size_t ws_size,
                              hipStream_t stream)
{
    const float*     x    = (const float*)d_in[0];
    const long long* eidx = (const long long*)d_in[1];
    const float*     W    = (const float*)d_in[2];
    const float*     b    = (const float*)d_in[3];

    const int n = in_sizes[0] / CH;            // N = 500000
    const int E = in_sizes[1] / 2;             // E == N
    const long long* row = eidx;
    const long long* col = eidx + E;

    float* ws_out = (float*)d_ws;
    int*   cnt    = (int*)((char*)d_ws + (size_t)n * CH * sizeof(float));

    // 1) dense projection via WMMA
    gc_gemm_wmma<<<(n + 15) / 16, 256, 0, stream>>>(x, W, b, ws_out, n);

    // 2) histogram of row indices
    gc_zero_cnt<<<(n + 255) / 256, 256, 0, stream>>>(cnt, n);
    gc_hist<<<(E + 255) / 256, 256, 0, stream>>>(row, cnt, E);

    // 3) gather + combine
    const long long total = (long long)n * CH;
    gc_combine<<<(int)((total + 255) / 256), 256, 0, stream>>>(
        ws_out, col, cnt, (float*)d_out, n);
}
